// DWTConditioned2EntropyLayerZTBlock_8031588843697
// MI455X (gfx1250) — compile-verified, matrix operand order checked by reading, not run on hardware
//
#include <hip/hip_runtime.h>
#include <hip/hip_bf16.h>

// CDNA5 / gfx1250, wave32. Conv chains via v_wmma_f32_16x16x32_f16.

typedef __attribute__((ext_vector_type(16))) _Float16 v16h;
typedef __attribute__((ext_vector_type(8)))  _Float16 v8h;
typedef __attribute__((ext_vector_type(8)))  float    v8f;

union V16H { v16h v; _Float16 h[16]; };
union V8H  { v8h  v; _Float16 h[8];  };
union V8F  { v8f  v; float    f[8];  };

#define WB() __builtin_amdgcn_wave_barrier()

__device__ __forceinline__ float lrelu(float x)     { return x >= 0.f ? x : 0.01f * x; }
__device__ __forceinline__ float softplusf(float x) { return log1pf(expf(x)); }
__device__ __forceinline__ float sigmoidf(float x)  { return 1.f / (1.f + expf(-x)); }

// ---------------------------------------------------------------------------
// Weight pre-pack kernels: lay weights out as ready-to-load WMMA A-fragments.
// A-frag (16-bit, 16x32): lane L holds M=L%16, K-half=L/16, 16 contiguous
// halves (t=0..15 -> K = Khalf*16 + t).  Fragment = 32 lanes * 16 halves.
// ---------------------------------------------------------------------------
__global__ void prep_w1(const float* __restrict__ w1_1, const float* __restrict__ w1_2,
                        const float* __restrict__ w1_3, const float* __restrict__ w1_4,
                        _Float16* __restrict__ dst) {
  int idx = blockIdx.x * 256 + threadIdx.x;           // 98304 total
  if (idx >= 98304) return;
  int t = idx & 15, lane = (idx >> 4) & 31;
  int q = idx >> 9;
  int s = q & 1, mtile = (q >> 1) & 1, cw = q >> 2;   // cw = k*12 + n
  int n = cw % 12, k = cw / 12;
  int cink = k + 1;
  int cout = mtile * 16 + (lane & 15);
  int kk = s * 32 + (lane >> 4) * 16 + t;             // K = cin*9 + tap, padded to 64
  float v = 0.f;
  if (kk < cink * 9) {
    const float* w = (k == 0) ? w1_1 : (k == 1) ? w1_2 : (k == 2) ? w1_3 : w1_4;
    v = w[((n * 32 + cout) * cink + kk / 9) * 9 + (kk % 9)];
  }
  dst[idx] = (_Float16)v;
}

__global__ void prep_w2(const float* __restrict__ w2, _Float16* __restrict__ dst) {
  int idx = blockIdx.x * 256 + threadIdx.x;           // 442368 total
  if (idx >= 442368) return;
  int t = idx & 15, lane = (idx >> 4) & 31;
  int q = idx >> 9;
  int s = q % 9, mtile = (q / 9) & 1, cw = q / 18;    // K-step s = tap
  int cout = mtile * 16 + (lane & 15);
  int cin  = (lane >> 4) * 16 + t;
  int ky = s / 3, kx = s % 3;
  dst[idx] = (_Float16)w2[(((size_t)(cw * 32 + cout) * 32 + cin) * 3 + ky) * 3 + kx];
}

__global__ void prep_w34(const float* __restrict__ w, _Float16* __restrict__ dst) {
  int idx = blockIdx.x * 256 + threadIdx.x;           // 49152 total
  if (idx >= 49152) return;
  int t = idx & 15, lane = (idx >> 4) & 31;
  int q = idx >> 9;
  int mtile = q & 1, cw = q >> 1;
  int cout = mtile * 16 + (lane & 15);
  int cin  = (lane >> 4) * 16 + t;
  dst[idx] = (_Float16)w[(size_t)(cw * 32 + cout) * 32 + cin];
}

// ---------------------------------------------------------------------------
// Elementwise round (jnp.round == rint, RNE)
// ---------------------------------------------------------------------------
__global__ void round_kernel(const float* __restrict__ x, float* __restrict__ y, int n) {
  int i = blockIdx.x * 256 + threadIdx.x;
  if (i < n) y[i] = rintf(x[i]);
}

// ---------------------------------------------------------------------------
// Entropy-bottleneck pointwise MLP (width 3), produces quantized out + -log2 p
// ---------------------------------------------------------------------------
__global__ void eb_kernel(const float* __restrict__ x,
                          const float* __restrict__ m0,  const float* __restrict__ m13,
                          const float* __restrict__ m4,  const float* __restrict__ b03,
                          const float* __restrict__ b4,  const float* __restrict__ f,
                          const float* __restrict__ q,
                          float* __restrict__ out_q, float* __restrict__ out_si,
                          int C, int HW, int total) {
  int idx = blockIdx.x * 256 + threadIdx.x;
  if (idx >= total) return;
  int c = (idx / HW) % C;
  float med = q[c * 3 + 1];
  float v = x[idx];
  float o = rintf(v - med) + med;

  float lg[2];
  #pragma unroll
  for (int e = 0; e < 2; ++e) {
    float u = o + (e ? 0.5f : -0.5f);
    float xv[3], yv[3];
    #pragma unroll
    for (int r = 0; r < 3; ++r) {
      float t0 = softplusf(m0[c * 3 + r]) * u + b03[(0 * C + c) * 3 + r];
      xv[r] = t0 + tanhf(f[(0 * C + c) * 3 + r]) * tanhf(t0);
    }
    #pragma unroll
    for (int i = 0; i < 3; ++i) {
      #pragma unroll
      for (int r = 0; r < 3; ++r) {
        float t0 = b03[((i + 1) * C + c) * 3 + r];
        #pragma unroll
        for (int col = 0; col < 3; ++col)
          t0 += softplusf(m13[((i * C + c) * 3 + r) * 3 + col]) * xv[col];
        yv[r] = t0 + tanhf(f[((i + 1) * C + c) * 3 + r]) * tanhf(t0);
      }
      xv[0] = yv[0]; xv[1] = yv[1]; xv[2] = yv[2];
    }
    float t0 = b4[c];
    #pragma unroll
    for (int col = 0; col < 3; ++col) t0 += softplusf(m4[c * 3 + col]) * xv[col];
    lg[e] = t0;
  }
  float lo = lg[0], up = lg[1];
  float s = lo + up;
  float sign = (s > 0.f) ? -1.f : (s < 0.f ? 1.f : 0.f);
  float lik = fabsf(sigmoidf(sign * up) - sigmoidf(sign * lo));
  lik = fmaxf(lik, 1e-9f);
  out_q[idx]  = o;
  out_si[idx] = -log2f(lik);
}

// ---------------------------------------------------------------------------
// Fused conv-chain + Gaussian likelihood kernel.
// One wave = one 16-pixel strip at half-res (b, j, kpos, y, sx); runs both
// role chains (mu then sigma) and writes -log2(lik) directly.
// All LDS B-fragments are contiguous, aligned vector loads (no predication):
// out-of-range K columns carry finite garbage that multiplies zeroed A
// weights; out-of-range positions are clamped and their D columns dropped.
// Epilogues are branch-free (image-border handled by a multiplicative mask);
// only the final packed v8h LDS store is predicated.
// ---------------------------------------------------------------------------
struct YAB { _Float16 yA[16][32]; _Float16 yB[16][32]; };
union DepOrY {
  _Float16 dep[4][5][20];   // conv1 input tile (only live before im2col build)
  YAB s;                    // conv2/3/4 activations (live after)
};
struct alignas(32) WaveScratch {
  _Float16 im1[54][64];     // conv1 im2col: [position 3x18][K padded to 64]
  _Float16 y1[3][18][32];   // conv1 output strip, channel-contiguous
  DepOrY u;
};

__launch_bounds__(128)
__global__ void chain_kernel(const float* __restrict__ xfull,   // raw x, (B,3,2H,2W)
                             const float* __restrict__ qfull,   // round(x) full-res
                             const float* __restrict__ con,     // (B,3,H,W)
                             float* __restrict__ si_out,        // (B,3,2H,2W)
                             const _Float16* __restrict__ p1,
                             const _Float16* __restrict__ p2,
                             const _Float16* __restrict__ p3,
                             const _Float16* __restrict__ p4,
                             const float* __restrict__ b1g, const float* __restrict__ b2g,
                             const float* __restrict__ b3g, const float* __restrict__ b4g,
                             const float* __restrict__ w5g, const float* __restrict__ b5g,
                             int H, int W, int lvl) {
  __shared__ WaveScratch sm[4];
  const int wave = threadIdx.x >> 5;
  const int lane = threadIdx.x & 31;
  WaveScratch& S = sm[wave];

  const int sxn = W >> 4;
  long task = (long)blockIdx.x * 4 + wave;
  int sx = (int)(task % sxn);  long t1 = task / sxn;
  int y    = (int)(t1 % H);    t1 /= H;
  int kpos = (int)(t1 % 4);    t1 /= 4;
  int j    = (int)(t1 % 3);
  int b    = (int)(t1 / 3);

  const int cink = kpos + 1;
  const int H2 = H * 2, W2 = W * 2;
  const int khalf = lane >> 4;
  const int pix   = lane & 15;

  // ---- stage dep tile (cink channels, 5x20 with zero halo) ----
  {
    const float* conp = con   + (size_t)(b * 3 + j) * H * W;
    const float* qfp  = qfull + (size_t)(b * 3 + j) * H2 * W2;
    for (int idx = lane; idx < cink * 100; idx += 32) {
      int cin = idx / 100, rem = idx % 100;
      int rr = rem / 20, cc = rem % 20;
      int gy = y - 2 + rr;
      int gx = sx * 16 - 2 + cc;
      float v = 0.f;
      if (gy >= 0 && gy < H && gx >= 0 && gx < W) {
        if (cin == 0)      v = conp[(size_t)gy * W + gx];
        else if (cin == 1) v = qfp[(size_t)(2 * gy) * W2 + 2 * gx];
        else if (cin == 2) v = qfp[(size_t)(2 * gy) * W2 + 2 * gx + 1];
        else               v = qfp[(size_t)(2 * gy + 1) * W2 + 2 * gx];
      }
      S.u.dep[cin][rr][cc] = (_Float16)v;
    }
  }
  WB();

  // ---- build conv1 im2col once (role-independent). K >= cink*9 reads a
  //      clamped (finite) dep value; the matching A weights are zero. ----
  for (int chunk = lane; chunk < 54 * 8; chunk += 32) {
    int p   = chunk >> 3;
    int kk0 = (chunk & 7) * 8;
    int r0 = p / 18, c0 = p % 18;
    V8H pk;
    #pragma unroll
    for (int e = 0; e < 8; ++e) {
      int kk = kk0 + e;
      int cin = kk / 9; if (cin >= cink) cin = 0;   // finite dummy, A weight is 0
      int tap = kk % 9;
      pk.h[e] = S.u.dep[cin][r0 + tap / 3][c0 + tap % 3];
    }
    *(v8h*)&S.im1[p][kk0] = pk.v;
  }
  WB();

  float mu_reg = 0.f;

  for (int role = 0; role < 2; ++role) {
    const int n  = (lvl * 3 + j) * 2 + role;
    const int cw = kpos * 12 + n;

    // ---- conv1: 3x3, cink->32, on the 3x18 strip conv2 needs ----
    for (int mtile = 0; mtile < 2; ++mtile) {
      v16h a0 = *(const v16h*)(p1 + ((size_t)((cw * 2 + mtile) * 2 + 0) * 32 + lane) * 16);
      v16h a1 = *(const v16h*)(p1 + ((size_t)((cw * 2 + mtile) * 2 + 1) * 32 + lane) * 16);
      const float4* bp = (const float4*)(b1g + cw * 32 + mtile * 16 + khalf * 8);
      float4 bv0 = bp[0], bv1 = bp[1];
      const float bb[8] = { bv0.x, bv0.y, bv0.z, bv0.w, bv1.x, bv1.y, bv1.z, bv1.w };
      for (int ntile = 0; ntile < 4; ++ntile) {
        int p  = ntile * 16 + pix;
        int pc = p < 54 ? p : 53;                    // clamp, finite garbage
        v16h bA = *(const v16h*)&S.im1[pc][0  + khalf * 16];
        v16h bB = *(const v16h*)&S.im1[pc][32 + khalf * 16];
        V8F acc;
        #pragma unroll
        for (int r = 0; r < 8; ++r) acc.f[r] = 0.f;
        acc.v = __builtin_amdgcn_wmma_f32_16x16x32_f16(false, a0, false, bA,
                                                       (short)0, acc.v, false, false);
        acc.v = __builtin_amdgcn_wmma_f32_16x16x32_f16(false, a1, false, bB,
                                                       (short)0, acc.v, false, false);
        // branch-free epilogue: image-border zeroing via multiplicative mask
        int r0 = pc / 18, c0 = pc % 18;
        int gy = y - 1 + r0;
        int gx = sx * 16 - 1 + c0;
        float msk = (gy >= 0 && gy < H && gx >= 0 && gx < W) ? 1.f : 0.f;
        V8H o;
        #pragma unroll
        for (int r = 0; r < 8; ++r)
          o.h[r] = (_Float16)(lrelu(acc.f[r] + bb[r]) * msk);
        if (p < 54)
          *(v8h*)&S.y1[r0][c0][mtile * 16 + khalf * 8] = o.v;
      }
    }
    WB();

    // ---- conv2: 3x3, 32->32, K=288 as 9 tap-major K=32 steps ----
    for (int mtile = 0; mtile < 2; ++mtile) {
      V8F acc;
      #pragma unroll
      for (int r = 0; r < 8; ++r) acc.f[r] = 0.f;
      #pragma unroll
      for (int s = 0; s < 9; ++s) {
        v16h a = *(const v16h*)(p2 + ((size_t)((cw * 2 + mtile) * 9 + s) * 32 + lane) * 16);
        v16h bb = *(const v16h*)&S.y1[s / 3][pix + (s % 3)][khalf * 16];
        acc.v = __builtin_amdgcn_wmma_f32_16x16x32_f16(false, a, false, bb,
                                                       (short)0, acc.v, false, false);
      }
      const float4* bp = (const float4*)(b2g + cw * 32 + mtile * 16 + khalf * 8);
      float4 bv0 = bp[0], bv1 = bp[1];
      const float bb2[8] = { bv0.x, bv0.y, bv0.z, bv0.w, bv1.x, bv1.y, bv1.z, bv1.w };
      V8H o;
      #pragma unroll
      for (int r = 0; r < 8; ++r) o.h[r] = (_Float16)lrelu(acc.f[r] + bb2[r]);
      *(v8h*)&S.u.s.yA[pix][mtile * 16 + khalf * 8] = o.v;
    }
    WB();

    // ---- conv3: 1x1, 32->32 ----
    for (int mtile = 0; mtile < 2; ++mtile) {
      V8F acc;
      #pragma unroll
      for (int r = 0; r < 8; ++r) acc.f[r] = 0.f;
      v16h a  = *(const v16h*)(p3 + ((size_t)(cw * 2 + mtile) * 32 + lane) * 16);
      v16h bb = *(const v16h*)&S.u.s.yA[pix][khalf * 16];
      acc.v = __builtin_amdgcn_wmma_f32_16x16x32_f16(false, a, false, bb,
                                                     (short)0, acc.v, false, false);
      const float4* bp = (const float4*)(b3g + cw * 32 + mtile * 16 + khalf * 8);
      float4 bv0 = bp[0], bv1 = bp[1];
      const float bb3[8] = { bv0.x, bv0.y, bv0.z, bv0.w, bv1.x, bv1.y, bv1.z, bv1.w };
      V8H o;
      #pragma unroll
      for (int r = 0; r < 8; ++r) o.h[r] = (_Float16)lrelu(acc.f[r] + bb3[r]);
      *(v8h*)&S.u.s.yB[pix][mtile * 16 + khalf * 8] = o.v;
    }
    WB();

    // ---- conv4: 1x1, 32->32 ----
    for (int mtile = 0; mtile < 2; ++mtile) {
      V8F acc;
      #pragma unroll
      for (int r = 0; r < 8; ++r) acc.f[r] = 0.f;
      v16h a  = *(const v16h*)(p4 + ((size_t)(cw * 2 + mtile) * 32 + lane) * 16);
      v16h bb = *(const v16h*)&S.u.s.yB[pix][khalf * 16];
      acc.v = __builtin_amdgcn_wmma_f32_16x16x32_f16(false, a, false, bb,
                                                     (short)0, acc.v, false, false);
      const float4* bp = (const float4*)(b4g + cw * 32 + mtile * 16 + khalf * 8);
      float4 bv0 = bp[0], bv1 = bp[1];
      const float bb4[8] = { bv0.x, bv0.y, bv0.z, bv0.w, bv1.x, bv1.y, bv1.z, bv1.w };
      V8H o;
      #pragma unroll
      for (int r = 0; r < 8; ++r) o.h[r] = (_Float16)lrelu(acc.f[r] + bb4[r]);
      *(v8h*)&S.u.s.yA[pix][mtile * 16 + khalf * 8] = o.v;
    }
    WB();

    // ---- conv5 (32->1) + likelihood epilogue on lanes 0..15 ----
    if (lane < 16) {
      V16H r0u, r1u;
      r0u.v = *(const v16h*)&S.u.s.yA[lane][0];
      r1u.v = *(const v16h*)&S.u.s.yA[lane][16];
      float acc = b5g[cw];
      #pragma unroll
      for (int ch = 0; ch < 16; ++ch) {
        acc += w5g[cw * 32 + ch]      * (float)r0u.h[ch];
        acc += w5g[cw * 32 + 16 + ch] * (float)r1u.h[ch];
      }
      if (role == 0) {
        mu_reg = acc;               // mu kept in registers; sigma chain next
      } else {
        int gy2 = (kpos >> 1) + 2 * y;
        int gx2 = (kpos & 1) + 2 * (sx * 16 + lane);
        size_t o = ((size_t)(b * 3 + j) * H2 + gy2) * W2 + gx2;
        float m = mu_reg;
        float xv = xfull[o];
        float outq = rintf(xv - m) + m;
        float sc = fmaxf(acc, 0.11f);
        float vv = fabsf(outq - m);
        const float inv_s2 = 0.70710678118654752f;
        float up = 0.5f * erfcf(((vv - 0.5f) / sc) * inv_s2);
        float lo = 0.5f * erfcf(((vv + 0.5f) / sc) * inv_s2);
        float lik = fmaxf(up - lo, 1e-9f);
        si_out[o] = -log2f(lik);
      }
    }
    WB();
  }
}

// ---------------------------------------------------------------------------
extern "C" void kernel_launch(void* const* d_in, const int* in_sizes, int n_in,
                              void* d_out, int out_size, void* d_ws, size_t ws_size,
                              hipStream_t stream) {
  (void)in_sizes; (void)n_in; (void)out_size; (void)ws_size;

  const float* out_xe = (const float*)d_in[0];
  const float* xo0    = (const float*)d_in[1];
  const float* xo1    = (const float*)d_in[2];
  const float* xo2    = (const float*)d_in[3];
  const float* w1_1   = (const float*)d_in[4];
  const float* w1_2   = (const float*)d_in[5];
  const float* w1_3   = (const float*)d_in[6];
  const float* w1_4   = (const float*)d_in[7];
  const float* b1     = (const float*)d_in[8];
  const float* w2     = (const float*)d_in[9];
  const float* b2     = (const float*)d_in[10];
  const float* w3     = (const float*)d_in[11];
  const float* b3     = (const float*)d_in[12];
  const float* w4     = (const float*)d_in[13];
  const float* b4     = (const float*)d_in[14];
  const float* w5     = (const float*)d_in[15];
  const float* b5     = (const float*)d_in[16];

  float* out = (float*)d_out;
  // d_out layout (floats, return order)
  const size_t o_si_xe = 0;
  const size_t o_si0 = o_si_xe + 131072;      // (8,3,512,512)
  const size_t o_si1 = o_si0 + 6291456;       // (8,3,256,256)
  const size_t o_si2 = o_si1 + 1572864;       // (8,3,128,128)
  const size_t o_xeq = o_si2 + 393216;        // (8,1,128,128)
  const size_t o_q0  = o_xeq + 131072;
  const size_t o_q1  = o_q0 + 6291456;
  const size_t o_q2  = o_q1 + 1572864;

  // packed f16 weight fragments in workspace
  _Float16* p1 = (_Float16*)d_ws;
  _Float16* p2 = p1 + 98304;
  _Float16* p3 = p2 + 442368;
  _Float16* p4 = p3 + 49152;

  prep_w1 <<<384,  256, 0, stream>>>(w1_1, w1_2, w1_3, w1_4, p1);
  prep_w2 <<<1728, 256, 0, stream>>>(w2, p2);
  prep_w34<<<192,  256, 0, stream>>>(w3, p3);
  prep_w34<<<192,  256, 0, stream>>>(w4, p4);

  round_kernel<<<24576, 256, 0, stream>>>(xo0, out + o_q0, 6291456);
  round_kernel<<<6144,  256, 0, stream>>>(xo1, out + o_q1, 1572864);

  eb_kernel<<<512, 256, 0, stream>>>(out_xe,
      (const float*)d_in[17], (const float*)d_in[18], (const float*)d_in[19],
      (const float*)d_in[20], (const float*)d_in[21], (const float*)d_in[22],
      (const float*)d_in[23], out + o_xeq, out + o_si_xe, 1, 128 * 128, 131072);
  eb_kernel<<<1536, 256, 0, stream>>>(xo2,
      (const float*)d_in[24], (const float*)d_in[25], (const float*)d_in[26],
      (const float*)d_in[27], (const float*)d_in[28], (const float*)d_in[29],
      (const float*)d_in[30], out + o_q2, out + o_si2, 3, 128 * 128, 393216);

  // level 0: x=xo1 (256^2), con=qnt(xo2), deps from round(xo1); 98304 wave-tasks
  chain_kernel<<<24576, 128, 0, stream>>>(xo1, out + o_q1, out + o_q2, out + o_si1,
      p1, p2, p3, p4, b1, b2, b3, b4, w5, b5, 128, 128, 0);
  // level 1: x=xo0 (512^2), con=round(xo1), deps from round(xo0); 393216 wave-tasks
  chain_kernel<<<98304, 128, 0, stream>>>(xo0, out + o_q0, out + o_q1, out + o_si0,
      p1, p2, p3, p4, b1, b2, b3, b4, w5, b5, 256, 256, 1);
}
